// MambaNet_9242769621279
// MI455X (gfx1250) — compile-verified, matrix operand order checked by reading, not run on hardware
//
#include <hip/hip_runtime.h>
#include <hip/hip_bf16.h>

// ---------------------------------------------------------------------------
// MambaNet (B=512, L=1) for gfx1250: f16-input / f32-accum WMMA GEMM pipeline.
// L=1 => scan collapses: y[b,c] = dt[b,c]*xin[b,c]*dot(Bm[b],Cm[b]); A_log dead.
// Register-blocked: each wave computes NT consecutive 16x16 N-tiles sharing
// one A fragment (NT=4 for big GEMMs) -> 1.25 fragment loads per WMMA.
// ---------------------------------------------------------------------------

typedef __attribute__((ext_vector_type(16))) _Float16 v16h;
typedef __attribute__((ext_vector_type(8)))  _Float16 v8h;
typedef __attribute__((ext_vector_type(8)))  float    v8f;

#define DEVINL __device__ __forceinline__

// Load one 16-half A/B fragment slice for this lane: two contiguous runs of 8
// halfs (16B each), per CDNA5 16-bit A-matrix 16x32 layout.
DEVINL v16h ld_frag(const _Float16* p) {
  union { v8h h[2]; v16h v; } u;
  u.h[0] = *(const v8h*)(p);
  u.h[1] = *(const v8h*)(p + 16);
  return u.v;
}

// acc[j] += A[m0:m0+16, :K] * W[n0+16j : n0+16j+16, :K]^T  for j in [0,NT)
// (W stored N x K row-major; one shared A fragment per K-step)
template <int NT>
DEVINL void gemm_tileN(const _Float16* __restrict__ A, int lda,
                       const _Float16* __restrict__ W, int ldw,
                       int m0, int n0, int K, int lane, v8f* acc) {
  const _Float16* pa = A + (size_t)(m0 + (lane & 15)) * lda + ((lane >> 4) << 3);
  const _Float16* pw = W + (size_t)(n0 + (lane & 15)) * ldw + ((lane >> 4) << 3);
  for (int k = 0; k < K; k += 32) {
    v16h a = ld_frag(pa + k);
#pragma unroll
    for (int j = 0; j < NT; ++j) {
      v16h b = ld_frag(pw + (size_t)j * 16 * ldw + k);
      acc[j] = __builtin_amdgcn_wmma_f32_16x16x32_f16(false, a, false, b,
                                                      (short)0, acc[j],
                                                      false, false);
    }
  }
}

DEVINL float silu_f(float x)     { return x / (1.f + __expf(-x)); }
DEVINL float softplus_f(float x) { return (x > 20.f) ? x : log1pf(__expf(x)); }

// ---------------------------------------------------------------------------
__global__ void k_cvt_f32_f16(const float* __restrict__ s,
                              _Float16* __restrict__ d, int n) {
  int i = blockIdx.x * blockDim.x + threadIdx.x;
  if (i < n) d[i] = (_Float16)s[i];
}

// xz = x @ in_proj_w^T  (M=512,K=512,N=2048); fused conv(k=1 tap)+silu split.
// NT=4 supertiles of 64 cols; 64 | 1024 so the xin/z split is wave-uniform.
__global__ void k_inproj(const _Float16* __restrict__ xh,
                         const _Float16* __restrict__ wh,
                         const float* __restrict__ conv_w,
                         const float* __restrict__ conv_b,
                         _Float16* __restrict__ xin_h,
                         float* __restrict__ sz) {
  int lane = threadIdx.x & 31, wv = threadIdx.x >> 5;
  int tile = blockIdx.x * 8 + wv;          // 32x32 supertiles
  int tn = tile % 32, tm = tile / 32;
  int m0 = tm * 16, n0 = tn * 64;
  v8f acc[4] = {};
  gemm_tileN<4>(xh, 512, wh, 512, m0, n0, 512, lane, acc);
  int mb = m0 + ((lane >> 4) << 3);
  if (n0 < 1024) {
#pragma unroll
    for (int j = 0; j < 4; ++j) {
      int n = n0 + j * 16 + (lane & 15);
      float cw = conv_w[n * 2 + 1];        // k=0 tap hits the zero pad (L=1)
      float cb = conv_b[n];
      for (int r = 0; r < 8; ++r)
        xin_h[(size_t)(mb + r) * 1024 + n] =
            (_Float16)silu_f(cb + cw * acc[j][r]);
    }
  } else {
#pragma unroll
    for (int j = 0; j < 4; ++j) {
      int nz = n0 - 1024 + j * 16 + (lane & 15);
      for (int r = 0; r < 8; ++r)
        sz[(size_t)(mb + r) * 1024 + nz] = silu_f(acc[j][r]);
    }
  }
}

// x_dbl = xin @ x_proj_w^T  (M=512,K=1024,N=288); NT=2 (18 tiles -> 9 pairs).
__global__ void k_xproj(const _Float16* __restrict__ xin_h,
                        const _Float16* __restrict__ wxp,
                        float* __restrict__ xdbl,
                        _Float16* __restrict__ dtraw) {
  int lane = threadIdx.x & 31, wv = threadIdx.x >> 5;
  int tile = blockIdx.x * 8 + wv;          // 32x9 supertiles
  int tn = tile % 9, tm = tile / 9;
  int m0 = tm * 16, n0 = tn * 32;
  v8f acc[2] = {};
  gemm_tileN<2>(xin_h, 1024, wxp, 1024, m0, n0, 1024, lane, acc);
  int mb = m0 + ((lane >> 4) << 3);
#pragma unroll
  for (int j = 0; j < 2; ++j) {
    int n = n0 + j * 16 + (lane & 15);
    for (int r = 0; r < 8; ++r) {
      xdbl[(size_t)(mb + r) * 288 + n] = acc[j][r];
      if (n < 32) dtraw[(size_t)(mb + r) * 32 + n] = (_Float16)acc[j][r];
    }
  }
}

// BC[b] = dot(Bm[b], Cm[b])   (one wave per row)
__global__ void k_bc(const float* __restrict__ xdbl, float* __restrict__ bc) {
  int lane = threadIdx.x & 31, wv = threadIdx.x >> 5;
  int b = blockIdx.x * 8 + wv;
  const float* row = xdbl + (size_t)b * 288;
  float s = 0.f;
  for (int i = lane; i < 128; i += 32) s += row[32 + i] * row[160 + i];
  for (int off = 16; off; off >>= 1) s += __shfl_xor(s, off, 32);
  if (lane == 0) bc[b] = s;
}

// dt = softplus(dt_raw @ dt_proj_w^T + b); y = (dt*xin*BC + D*xin)*silu(z)
// K=32: one WMMA per N-tile; NT=4 amortizes the single A fragment 4x.
__global__ void k_dt(const _Float16* __restrict__ dtraw,
                     const _Float16* __restrict__ wdt,
                     const float* __restrict__ dtb,
                     const float* __restrict__ bc,
                     const _Float16* __restrict__ xin_h,
                     const float* __restrict__ Dp,
                     const float* __restrict__ sz,
                     _Float16* __restrict__ yh) {
  int lane = threadIdx.x & 31, wv = threadIdx.x >> 5;
  int tile = blockIdx.x * 8 + wv;          // 32x16 supertiles
  int tn = tile % 16, tm = tile / 16;
  int m0 = tm * 16, n0 = tn * 64;
  v8f acc[4] = {};
  gemm_tileN<4>(dtraw, 32, wdt, 32, m0, n0, 32, lane, acc);
  int mb = m0 + ((lane >> 4) << 3);
#pragma unroll
  for (int j = 0; j < 4; ++j) {
    int n = n0 + j * 16 + (lane & 15);
    float bias = dtb[n], dd = Dp[n];
    for (int r = 0; r < 8; ++r) {
      int m = mb + r;
      float dt = softplus_f(acc[j][r] + bias);
      float xi = (float)xin_h[(size_t)m * 1024 + n];
      float y  = (dt * xi * bc[m] + dd * xi) * sz[(size_t)m * 1024 + n];
      yh[(size_t)m * 1024 + n] = (_Float16)y;
    }
  }
}

// out = y @ out_proj_w^T  (M=512,K=1024,N=512)
__global__ void k_outproj(const _Float16* __restrict__ yh,
                          const _Float16* __restrict__ wout,
                          _Float16* __restrict__ outh) {
  int lane = threadIdx.x & 31, wv = threadIdx.x >> 5;
  int tile = blockIdx.x * 8 + wv;          // 32x8 supertiles
  int tn = tile % 8, tm = tile / 8;
  int m0 = tm * 16, n0 = tn * 64;
  v8f acc[4] = {};
  gemm_tileN<4>(yh, 1024, wout, 1024, m0, n0, 1024, lane, acc);
  int mb = m0 + ((lane >> 4) << 3);
#pragma unroll
  for (int j = 0; j < 4; ++j) {
    int n = n0 + j * 16 + (lane & 15);
    for (int r = 0; r < 8; ++r)
      outh[(size_t)(mb + r) * 512 + n] = (_Float16)acc[j][r];
  }
}

// h1 = leaky_relu(out @ fc1_w^T + b, 0.1)  (M=512,K=512,N=256)
__global__ void k_fc1(const _Float16* __restrict__ outh,
                      const _Float16* __restrict__ w1,
                      const float* __restrict__ b1,
                      _Float16* __restrict__ h1) {
  int lane = threadIdx.x & 31, wv = threadIdx.x >> 5;
  int tile = blockIdx.x * 8 + wv;          // 32x4 supertiles
  int tn = tile % 4, tm = tile / 4;
  int m0 = tm * 16, n0 = tn * 64;
  v8f acc[4] = {};
  gemm_tileN<4>(outh, 512, w1, 512, m0, n0, 512, lane, acc);
  int mb = m0 + ((lane >> 4) << 3);
#pragma unroll
  for (int j = 0; j < 4; ++j) {
    int n = n0 + j * 16 + (lane & 15);
    float bias = b1[n];
    for (int r = 0; r < 8; ++r) {
      float h = acc[j][r] + bias;
      h1[(size_t)(mb + r) * 256 + n] = (_Float16)((h >= 0.f) ? h : 0.1f * h);
    }
  }
}

// o[b] = sigmoid(dot(h1[b], fc5_w) + fc5_b)   (one wave per row)
__global__ void k_fc5(const _Float16* __restrict__ h1,
                      const float* __restrict__ w5,
                      const float* __restrict__ b5,
                      float* __restrict__ out) {
  int lane = threadIdx.x & 31, wv = threadIdx.x >> 5;
  int b = blockIdx.x * 8 + wv;
  float s = 0.f;
  for (int i = lane; i < 256; i += 32)
    s += (float)h1[(size_t)b * 256 + i] * w5[i];
  for (int off = 16; off; off >>= 1) s += __shfl_xor(s, off, 32);
  if (lane == 0) out[b] = 1.f / (1.f + __expf(-(s + b5[0])));
}

// ---------------------------------------------------------------------------
extern "C" void kernel_launch(void* const* d_in, const int* in_sizes, int n_in,
                              void* d_out, int out_size, void* d_ws, size_t ws_size,
                              hipStream_t stream) {
  const float* x         = (const float*)d_in[0];   // 512*512
  const float* in_proj_w = (const float*)d_in[1];   // 2048*512
  const float* conv_w    = (const float*)d_in[2];   // 1024*2
  const float* conv_b    = (const float*)d_in[3];   // 1024
  const float* x_proj_w  = (const float*)d_in[4];   // 288*1024
  const float* dt_proj_w = (const float*)d_in[5];   // 1024*32
  const float* dt_proj_b = (const float*)d_in[6];   // 1024
  // d_in[7] = A_log (dead for L=1 with h0=0)
  const float* Dp        = (const float*)d_in[8];   // 1024
  const float* out_projw = (const float*)d_in[9];   // 512*1024
  const float* fc1_w     = (const float*)d_in[10];  // 256*512
  const float* fc1_b     = (const float*)d_in[11];  // 256
  const float* fc5_w     = (const float*)d_in[12];  // 256
  const float* fc5_b     = (const float*)d_in[13];  // 1
  float* out = (float*)d_out;                       // 512

  char* ws = (char*)d_ws;
  size_t off = 0;
  auto alloc = [&](size_t bytes) -> char* {
    char* p = ws + off;
    off += (bytes + 255) & ~(size_t)255;
    return p;
  };
  _Float16* x_h     = (_Float16*)alloc(512u * 512 * 2);
  _Float16* w_in_h  = (_Float16*)alloc(2048u * 512 * 2);
  _Float16* xin_h   = (_Float16*)alloc(512u * 1024 * 2);
  float*    sz_f    = (float*)   alloc(512u * 1024 * 4);
  _Float16* w_xp_h  = (_Float16*)alloc(288u * 1024 * 2);
  float*    xdbl_f  = (float*)   alloc(512u * 288 * 4);
  _Float16* dtraw_h = (_Float16*)alloc(512u * 32 * 2);
  float*    bc_f    = (float*)   alloc(512u * 4);
  _Float16* w_dt_h  = (_Float16*)alloc(1024u * 32 * 2);
  _Float16* y_h     = (_Float16*)alloc(512u * 1024 * 2);
  _Float16* w_out_h = (_Float16*)alloc(512u * 1024 * 2);
  _Float16* out_h   = (_Float16*)alloc(512u * 512 * 2);
  _Float16* w_fc1_h = (_Float16*)alloc(256u * 512 * 2);
  _Float16* h1_h    = (_Float16*)alloc(512u * 256 * 2);

  const int T = 256;
  auto cvt = [&](const float* s, _Float16* d, int n) {
    k_cvt_f32_f16<<<(n + T - 1) / T, T, 0, stream>>>(s, d, n);
  };
  cvt(x,         x_h,     512 * 512);
  cvt(in_proj_w, w_in_h,  2048 * 512);
  cvt(x_proj_w,  w_xp_h,  288 * 1024);
  cvt(dt_proj_w, w_dt_h,  1024 * 32);
  cvt(out_projw, w_out_h, 512 * 1024);
  cvt(fc1_w,     w_fc1_h, 256 * 512);

  k_inproj <<<128, T, 0, stream>>>(x_h, w_in_h, conv_w, conv_b, xin_h, sz_f);
  k_xproj  <<<36,  T, 0, stream>>>(xin_h, w_xp_h, xdbl_f, dtraw_h);
  k_bc     <<<64,  T, 0, stream>>>(xdbl_f, bc_f);
  k_dt     <<<64,  T, 0, stream>>>(dtraw_h, w_dt_h, dt_proj_b, bc_f,
                                   xin_h, Dp, sz_f, y_h);
  k_outproj<<<32,  T, 0, stream>>>(y_h, w_out_h, out_h);
  k_fc1    <<<16,  T, 0, stream>>>(out_h, w_fc1_h, fc1_b, h1_h);
  k_fc5    <<<64,  T, 0, stream>>>(h1_h, fc5_w, fc5_b, out);

  (void)in_sizes; (void)n_in; (void)out_size; (void)ws_size;
}
